// CFGEncoder_34720515621382
// MI455X (gfx1250) — matrix operand order repaired and checked by reading, not compiled, not run on hardware
//
#include <hip/hip_runtime.h>
#include <hip/hip_bf16.h>
#include <math.h>

#define LAYERS 3
#define HEADS  4
#define CH     128
#define DIM    128
#define HC     512      // HEADS*CH
#define NGRAPH 64
#define OUTD   128
#define NEG_SLOPE 0.2f
#define LN_EPS 1e-5f

typedef __attribute__((ext_vector_type(16))) __bf16 v16bf;
typedef __attribute__((ext_vector_type(8)))  float  v8f;

// ---- monotone float <-> uint mapping for atomic segment-max ----
__device__ __forceinline__ unsigned fkey(float f) {
  unsigned u = __float_as_uint(f);
  return (u & 0x80000000u) ? ~u : (u | 0x80000000u);
}
__device__ __forceinline__ float funkey(unsigned k) {
  unsigned b = (k & 0x80000000u) ? (k & 0x7FFFFFFFu) : ~k;
  return __uint_as_float(b);
}

// ---------------- utility kernels ----------------
__global__ void zero_f32(float* p, long n) {
  long i = (long)blockIdx.x * blockDim.x + threadIdx.x;
  if (i < n) p[i] = 0.f;
}
__global__ void zero_u32(unsigned* p, long n) {
  long i = (long)blockIdx.x * blockDim.x + threadIdx.x;
  if (i < n) p[i] = 0u;
}
__global__ void copy_f32(float* d, const float* s, long n) {
  long i = (long)blockIdx.x * blockDim.x + threadIdx.x;
  if (i < n) d[i] = s[i];
}
// split-precision: s = hi + lo with both parts representable in bf16
__global__ void f32_split_bf16(__bf16* hi, __bf16* lo, const float* s, long n) {
  long i = (long)blockIdx.x * blockDim.x + threadIdx.x;
  if (i < n) {
    const float v = s[i];
    const __bf16 h = (__bf16)v;
    hi[i] = h;
    lo[i] = (__bf16)(v - (float)h);
  }
}

// ---------------- WMMA projection GEMM (bf16x3 split ~= fp32) ----------------
// Y[N,512] = h[N,128] @ W[512,128]^T + bias, for Wl (z=0) and Wr (z=1).
// One wave computes one 16(row)x16(col) tile; K=128 -> 4 K-steps x 3 WMMAs:
//   acc += a_hi*b_hi + a_hi*b_lo + a_lo*b_hi   (fp32 accumulate)
__global__ void gemm_proj(const __bf16* __restrict__ hHi, const __bf16* __restrict__ hLo,
                          const __bf16* __restrict__ WlHi, const __bf16* __restrict__ WlLo,
                          const __bf16* __restrict__ WrHi, const __bf16* __restrict__ WrLo,
                          const float* __restrict__ bL,  const float* __restrict__ bR,
                          float* __restrict__ Yl, float* __restrict__ Yr, int N)
{
  const int lane = threadIdx.x & 31;
  const int wv   = threadIdx.x >> 5;                 // 8 waves per block
  const int colBase = (blockIdx.y * 8 + wv) * 16;    // 32 col tiles of 16
  const int rowBase = blockIdx.x * 16;

  const __bf16* Whi  = blockIdx.z ? WrHi : WlHi;
  const __bf16* Wlo  = blockIdx.z ? WrLo : WlLo;
  const float*  bias = blockIdx.z ? bR  : bL;
  float*        Y    = blockIdx.z ? Yr  : Yl;

  const int m    = lane & 15;
  const int kgrp = lane >> 4;                        // 0 or 1 (lane half)
  int arow = rowBase + m; if (arow >= N) arow = N - 1;
  const size_t hoff = (size_t)arow * DIM;
  const size_t woff = (size_t)(colBase + m) * DIM;   // B col n == lane&15

  v8f acc = {};
  #pragma unroll
  for (int ks = 0; ks < 4; ++ks) {
    const int kb = ks * 32;
    v16bf ah, al, bh, bl2;
    #pragma unroll
    for (int v = 0; v < 8; ++v) {
      // A 16x32 bf16 layout: lanes0-15 hold K {0..7,16..23}, lanes16-31 hold K {8..15,24..31}
      const int ka = kb + ((v < 4) ? 2 * v : 16 + 2 * (v - 4)) + kgrp * 8;
      ah[2 * v]     = hHi[hoff + ka];
      ah[2 * v + 1] = hHi[hoff + ka + 1];
      al[2 * v]     = hLo[hoff + ka];
      al[2 * v + 1] = hLo[hoff + ka + 1];
      // B 32x16 bf16 layout: same K order as A (lanes0-15 K {0..7,16..23}, lanes16-31 K {8..15,24..31})
      const int kp = kb + ((v < 4) ? 2 * v : 16 + 2 * (v - 4)) + kgrp * 8;
      bh[2 * v]      = Whi[woff + kp];
      bh[2 * v + 1]  = Whi[woff + kp + 1];
      bl2[2 * v]     = Wlo[woff + kp];
      bl2[2 * v + 1] = Wlo[woff + kp + 1];
    }
    acc = __builtin_amdgcn_wmma_f32_16x16x32_bf16(false, ah, false, bh,
                                                  (short)0, acc, false, false);
    acc = __builtin_amdgcn_wmma_f32_16x16x32_bf16(false, ah, false, bl2,
                                                  (short)0, acc, false, false);
    acc = __builtin_amdgcn_wmma_f32_16x16x32_bf16(false, al, false, bh,
                                                  (short)0, acc, false, false);
  }
  // D layout: VGPR r, lanes0-15 -> M=r, lanes16-31 -> M=8+r; N = lane&15
  const int n   = lane & 15;
  const int mof = (lane >> 4) * 8;
  const int col = colBase + n;
  const float bv = bias[col];
  if (rowBase + 16 <= N) {       // fast path: full tile, no guards
    #pragma unroll
    for (int r = 0; r < 8; ++r)
      Y[(size_t)(rowBase + mof + r) * HC + col] = acc[r] + bv;
  } else {
    #pragma unroll
    for (int r = 0; r < 8; ++r) {
      const int row = rowBase + mof + r;
      if (row < N) Y[(size_t)row * HC + col] = acc[r] + bv;
    }
  }
}

// ---------------- edge kernels (wave per (edge,head)) ----------------
__global__ void edge_logits(const float* __restrict__ xl, const float* __restrict__ xr,
                            const int* __restrict__ ei, const float* __restrict__ att,
                            float* __restrict__ logits, unsigned* __restrict__ mU,
                            int E, int N)
{
  long wid = (((long)blockIdx.x * blockDim.x) + threadIdx.x) >> 5;
  const int lane = threadIdx.x & 31;
  const long total = ((long)E + N) * HEADS;
  if (wid >= total) return;
  const long e = wid >> 2; const int hh = (int)(wid & 3);
  int src, dst;
  if (e < E) { src = ei[e]; dst = ei[E + e]; } else { src = dst = (int)(e - E); }

  const float4 l4 = ((const float4*)(xl + (size_t)src * HC + hh * CH))[lane];
  const float4 r4 = ((const float4*)(xr + (size_t)dst * HC + hh * CH))[lane];
  const float4 a4 = ((const float4*)(att + hh * CH))[lane];
  float s0 = l4.x + r4.x; s0 = s0 > 0.f ? s0 : NEG_SLOPE * s0;
  float s1 = l4.y + r4.y; s1 = s1 > 0.f ? s1 : NEG_SLOPE * s1;
  float s2 = l4.z + r4.z; s2 = s2 > 0.f ? s2 : NEG_SLOPE * s2;
  float s3 = l4.w + r4.w; s3 = s3 > 0.f ? s3 : NEG_SLOPE * s3;
  float acc = s0 * a4.x + s1 * a4.y + s2 * a4.z + s3 * a4.w;
  #pragma unroll
  for (int off = 16; off > 0; off >>= 1) acc += __shfl_down(acc, off, 32);
  if (lane == 0) {
    logits[e * HEADS + hh] = acc;
    atomicMax(&mU[(size_t)dst * HEADS + hh], fkey(acc));
  }
}

__global__ void edge_softmax_p(const float* __restrict__ logits, const int* __restrict__ ei,
                               const unsigned* __restrict__ mU,
                               float* __restrict__ p, float* __restrict__ denom,
                               int E, int N)
{
  long i = (long)blockIdx.x * blockDim.x + threadIdx.x;
  const long total = ((long)E + N) * HEADS;
  if (i >= total) return;
  const long e = i >> 2; const int hh = (int)(i & 3);
  const int dst = (e < E) ? ei[E + e] : (int)(e - E);
  const float m  = funkey(mU[(size_t)dst * HEADS + hh]);
  const float pv = expf(logits[i] - m);
  p[i] = pv;
  atomicAdd(&denom[(size_t)dst * HEADS + hh], pv);
}

__global__ void edge_aggregate(const float* __restrict__ xl, const int* __restrict__ ei,
                               const float* __restrict__ p, const float* __restrict__ denom,
                               float* __restrict__ agg, int E, int N)
{
  long wid = (((long)blockIdx.x * blockDim.x) + threadIdx.x) >> 5;
  const int lane = threadIdx.x & 31;
  const long total = ((long)E + N) * HEADS;
  if (wid >= total) return;
  const long e = wid >> 2; const int hh = (int)(wid & 3);
  int src, dst;
  if (e < E) { src = ei[e]; dst = ei[E + e]; } else { src = dst = (int)(e - E); }
  // fold mean-over-heads (1/H) into alpha
  const float alpha = p[e * HEADS + hh] / denom[(size_t)dst * HEADS + hh] * (1.f / HEADS);
  const float4 v = ((const float4*)(xl + (size_t)src * HC + hh * CH))[lane];
  float* o = agg + (size_t)dst * CH + lane * 4;
  atomicAdd(o + 0, v.x * alpha);
  atomicAdd(o + 1, v.y * alpha);
  atomicAdd(o + 2, v.z * alpha);
  atomicAdd(o + 3, v.w * alpha);
}

// ---------------- per-node LN + ReLU + residual ----------------
__global__ void node_update(float* __restrict__ h, const float* __restrict__ agg,
                            const float* __restrict__ cbias, const float* __restrict__ gamma,
                            const float* __restrict__ beta, int N)
{
  const int nidx = blockIdx.x;
  const int c = threadIdx.x;         // 128 threads
  __shared__ float red[CH];
  const float v = agg[(size_t)nidx * CH + c] + cbias[c];
  red[c] = v; __syncthreads();
  #pragma unroll
  for (int s = 64; s > 0; s >>= 1) { if (c < s) red[c] += red[c + s]; __syncthreads(); }
  const float mu = red[0] * (1.f / CH);
  __syncthreads();
  const float d = v - mu;
  red[c] = d * d; __syncthreads();
  #pragma unroll
  for (int s = 64; s > 0; s >>= 1) { if (c < s) red[c] += red[c + s]; __syncthreads(); }
  const float var = red[0] * (1.f / CH);
  float y = d * rsqrtf(var + LN_EPS) * gamma[c] + beta[c];
  y = fmaxf(y, 0.f);
  h[(size_t)nidx * CH + c] += y;
}

// ---------------- global mean pool ----------------
__global__ void pool_kernel(const float* __restrict__ h, const int* __restrict__ batch,
                            float* __restrict__ sums, float* __restrict__ cnt, int N)
{
  long i = (long)blockIdx.x * blockDim.x + threadIdx.x;
  const long total = (long)N * CH;
  if (i >= total) return;
  const int nidx = (int)(i >> 7);
  const int c = (int)(i & 127);
  const int b = batch[nidx];
  atomicAdd(&sums[(size_t)b * CH + c], h[i]);
  if (c == 0) atomicAdd(&cnt[b], 1.f);
}

// ---------------- MLP head + output LayerNorm (one block per graph) ----------------
__global__ void head_kernel(const float* __restrict__ sums, const float* __restrict__ cnt,
                            const float* __restrict__ W1, const float* __restrict__ b1,
                            const float* __restrict__ W2, const float* __restrict__ b2,
                            const float* __restrict__ og, const float* __restrict__ ob,
                            float* __restrict__ out)
{
  const int g = blockIdx.x;
  const int t = threadIdx.x;     // 256 threads
  __shared__ float gv[CH];
  __shared__ float hid[2 * CH];
  __shared__ float red[CH];
  if (t < CH) gv[t] = sums[(size_t)g * CH + t] / fmaxf(cnt[g], 1.f);
  __syncthreads();
  float a = b1[t];
  const float* w1 = W1 + (size_t)t * CH;
  #pragma unroll 4
  for (int d = 0; d < CH; ++d) a += gv[d] * w1[d];
  hid[t] = 0.5f * a * (1.f + erff(a * 0.70710678118654752f));   // exact gelu
  __syncthreads();
  float o = 0.f;
  if (t < CH) {
    o = b2[t];
    const float* w2 = W2 + (size_t)t * (2 * CH);
    #pragma unroll 4
    for (int j = 0; j < 2 * CH; ++j) o += hid[j] * w2[j];
    red[t] = o;
  }
  __syncthreads();
  #pragma unroll
  for (int s = 64; s > 0; s >>= 1) { if (t < s) red[t] += red[t + s]; __syncthreads(); }
  const float mu = red[0] * (1.f / CH);
  __syncthreads();
  const float dd = o - mu;
  if (t < CH) red[t] = dd * dd;
  __syncthreads();
  #pragma unroll
  for (int s = 64; s > 0; s >>= 1) { if (t < s) red[t] += red[t + s]; __syncthreads(); }
  const float var = red[0] * (1.f / CH);
  if (t < CH) out[(size_t)g * CH + t] = dd * rsqrtf(var + LN_EPS) * og[t] + ob[t];
}

// ---------------- launcher ----------------
extern "C" void kernel_launch(void* const* d_in, const int* in_sizes, int n_in,
                              void* d_out, int out_size, void* d_ws, size_t ws_size,
                              hipStream_t stream)
{
  const float* x   = (const float*)d_in[0];
  const int*   ei  = (const int*)d_in[1];
  const int*   bat = (const int*)d_in[2];
  const float* Wl  = (const float*)d_in[3];
  const float* bl  = (const float*)d_in[4];
  const float* Wr  = (const float*)d_in[5];
  const float* br  = (const float*)d_in[6];
  const float* att = (const float*)d_in[7];
  const float* cb  = (const float*)d_in[8];
  const float* lg  = (const float*)d_in[9];
  const float* lb  = (const float*)d_in[10];
  const float* W1  = (const float*)d_in[11];
  const float* b1  = (const float*)d_in[12];
  const float* W2  = (const float*)d_in[13];
  const float* b2  = (const float*)d_in[14];
  const float* og  = (const float*)d_in[15];
  const float* ob  = (const float*)d_in[16];

  const int N = in_sizes[0] / DIM;
  const int E = in_sizes[1] / 2;
  const long ET = (long)E + N;

  // workspace carve-out
  char* ws = (char*)d_ws;
  size_t off = 0;
  auto alloc = [&](size_t bytes) -> char* {
    off = (off + 255) & ~(size_t)255;
    char* p = ws + off; off += bytes; return p;
  };
  float*    hF    = (float*)   alloc((size_t)N * DIM * 4);
  __bf16*   hBhi  = (__bf16*)  alloc((size_t)N * DIM * 2);
  __bf16*   hBlo  = (__bf16*)  alloc((size_t)N * DIM * 2);
  __bf16*   WlBhi = (__bf16*)  alloc((size_t)LAYERS * HC * DIM * 2);
  __bf16*   WlBlo = (__bf16*)  alloc((size_t)LAYERS * HC * DIM * 2);
  __bf16*   WrBhi = (__bf16*)  alloc((size_t)LAYERS * HC * DIM * 2);
  __bf16*   WrBlo = (__bf16*)  alloc((size_t)LAYERS * HC * DIM * 2);
  float*    xlF   = (float*)   alloc((size_t)N * HC * 4);
  float*    xrF   = (float*)   alloc((size_t)N * HC * 4);
  float*    logi  = (float*)   alloc((size_t)ET * HEADS * 4);
  float*    pbuf  = (float*)   alloc((size_t)ET * HEADS * 4);
  unsigned* mU    = (unsigned*)alloc((size_t)N * HEADS * 4);
  float*    den   = (float*)   alloc((size_t)N * HEADS * 4);
  float*    agg   = (float*)   alloc((size_t)N * CH * 4);
  float*    psum  = (float*)   alloc((size_t)NGRAPH * CH * 4);
  float*    pcnt  = (float*)   alloc((size_t)NGRAPH * 4);
  (void)ws_size; (void)n_in; (void)out_size;

  const int TPB = 256;
  const long nhx = (long)N * DIM;
  copy_f32<<<(int)((nhx + TPB - 1) / TPB), TPB, 0, stream>>>(hF, x, nhx);
  const long nw = (long)LAYERS * HC * DIM;
  f32_split_bf16<<<(int)((nw + TPB - 1) / TPB), TPB, 0, stream>>>(WlBhi, WlBlo, Wl, nw);
  f32_split_bf16<<<(int)((nw + TPB - 1) / TPB), TPB, 0, stream>>>(WrBhi, WrBlo, Wr, nw);

  dim3 gGemm((N + 15) / 16, 4, 2);           // 8 waves/block -> 32 col tiles
  const long edgeWork = ET * HEADS;          // wave-level items
  const int  eBlkW = (int)((edgeWork * 32 + TPB - 1) / TPB);
  const int  eBlkT = (int)((edgeWork + TPB - 1) / TPB);
  const long nh4 = (long)N * HEADS;

  for (int l = 0; l < LAYERS; ++l) {
    f32_split_bf16<<<(int)((nhx + TPB - 1) / TPB), TPB, 0, stream>>>(hBhi, hBlo, hF, nhx);
    gemm_proj<<<gGemm, TPB, 0, stream>>>(hBhi, hBlo,
        WlBhi + (size_t)l * HC * DIM, WlBlo + (size_t)l * HC * DIM,
        WrBhi + (size_t)l * HC * DIM, WrBlo + (size_t)l * HC * DIM,
        bl + (size_t)l * HC, br + (size_t)l * HC, xlF, xrF, N);
    zero_u32<<<(int)((nh4 + TPB - 1) / TPB), TPB, 0, stream>>>(mU, nh4);
    zero_f32<<<(int)((nh4 + TPB - 1) / TPB), TPB, 0, stream>>>(den, nh4);
    zero_f32<<<(int)((nhx + TPB - 1) / TPB), TPB, 0, stream>>>(agg, nhx);
    edge_logits   <<<eBlkW, TPB, 0, stream>>>(xlF, xrF, ei, att + (size_t)l * HC, logi, mU, E, N);
    edge_softmax_p<<<eBlkT, TPB, 0, stream>>>(logi, ei, mU, pbuf, den, E, N);
    edge_aggregate<<<eBlkW, TPB, 0, stream>>>(xlF, ei, pbuf, den, agg, E, N);
    node_update<<<N, CH, 0, stream>>>(hF, agg,
        cb + (size_t)l * CH, lg + (size_t)l * CH, lb + (size_t)l * CH, N);
  }

  zero_f32<<<(NGRAPH * CH + TPB - 1) / TPB, TPB, 0, stream>>>(psum, NGRAPH * CH);
  zero_f32<<<1, TPB, 0, stream>>>(pcnt, NGRAPH);
  pool_kernel<<<(int)((nhx + TPB - 1) / TPB), TPB, 0, stream>>>(hF, bat, psum, pcnt, N);
  head_kernel<<<NGRAPH, 2 * CH, 0, stream>>>(psum, pcnt, W1, b1, W2, b2, og, ob, (float*)d_out);
}
